// HeteroGraphConv_72224170049980
// MI455X (gfx1250) — matrix operand order repaired and checked by reading, not compiled, not run on hardware
//
#include <hip/hip_runtime.h>
#include <hip/hip_bf16.h>

#define N_NODES 50000
#define N_EDGES 1600000
#define BATCH   16384
#define EMB     128
#define NH      4      // heads
#define HD      32     // feats per head (H*D = 128)
#define SLOPE   0.2f

typedef __attribute__((ext_vector_type(16))) __bf16 v16bf;
typedef __attribute__((ext_vector_type(8)))  float  v8f;

__device__ __forceinline__ void pack4(v16bf& v, int o, float4 f) {
  v[o + 0] = (__bf16)f.x; v[o + 1] = (__bf16)f.y;
  v[o + 2] = (__bf16)f.z; v[o + 3] = (__bf16)f.w;
}

// ---------------------------------------------------------------------------
// Generic WMMA bf16 GEMM:  C[m, col0+n] = act(A[m,:K] @ W[n,:K]^T + bias[n])
// One wave computes one 16x16 tile, K stepped by 32. Each lane's A/B fragment
// is two 8-wide K runs (ISA 7.12.2 layouts), so all loads are float4 (b128).
// Requires M%16==0, N%16==0, K%32==0 (true for every GEMM here).
// ---------------------------------------------------------------------------
__global__ __launch_bounds__(32)
void k_gemm_wmma(const float* __restrict__ A, const float* __restrict__ W,
                 const float* __restrict__ bias, float* __restrict__ C,
                 int K, int ldc, int col0, int relu) {
  const int tm   = blockIdx.x;       // M tile index
  const int tn   = blockIdx.y;       // N tile index
  const int lane = threadIdx.x;      // 0..31, wave32
  const int half = lane >> 4;        // 0 / 1
  const int l15  = lane & 15;

  // rows are 16B-aligned: K % 32 == 0
  const float4* A4 = (const float4*)(A + (size_t)(tm * 16 + l15) * K);
  const float4* W4 = (const float4*)(W + (size_t)(tn * 16 + l15) * K);

  v8f acc = {};
  for (int k0 = 0; k0 < K; k0 += 32) {
    // A 16x32 bf16 fragment: K = {k0+8h .. +7} U {k0+8h+16 .. +23}
    const int ia = (k0 + 8 * half) >> 2;
    float4 a0 = A4[ia],     a1 = A4[ia + 1];
    float4 a2 = A4[ia + 4], a3 = A4[ia + 5];
    // B 32x16 bf16 fragment: K = {k0+8h .. +7} U {k0+8h+16 .. +23}
    const int ib = (k0 + 8 * half) >> 2;
    float4 b0 = W4[ib],     b1 = W4[ib + 1];
    float4 b2 = W4[ib + 4], b3 = W4[ib + 5];

    v16bf af, bf;
    pack4(af, 0, a0);  pack4(af, 4, a1);  pack4(af, 8, a2);  pack4(af, 12, a3);
    pack4(bf, 0, b0);  pack4(bf, 4, b1);  pack4(bf, 8, b2);  pack4(bf, 12, b3);

    acc = __builtin_amdgcn_wmma_f32_16x16x32_bf16(
        /*neg_a=*/false, af, /*neg_b=*/false, bf,
        /*c_mod=*/(short)0, acc, /*reuse_a=*/false, /*reuse_b=*/false);
  }

  const int n  = tn * 16 + l15;
  const float bv = bias ? bias[n] : 0.0f;
#pragma unroll
  for (int v = 0; v < 8; ++v) {
    const int m = tm * 16 + v + 8 * half;   // C layout: VGPR v -> row v / v+8
    float val = acc[v] + bv;
    if (relu) val = fmaxf(val, 0.0f);
    C[(size_t)m * ldc + col0 + n] = val;
  }
}

// ---------------------------------------------------------------------------
// Embedding gather, float4-wide: out[i, :] = table[ids[i], :]
// ---------------------------------------------------------------------------
__global__ void k_gather(const float4* __restrict__ table,
                         const int* __restrict__ ids,
                         float4* __restrict__ out, int rows) {
  int idx = blockIdx.x * blockDim.x + threadIdx.x;
  if (idx >= rows * (EMB / 4)) return;
  int r = idx >> 5, c = idx & 31;
  out[idx] = table[(size_t)ids[r] * 32 + c];
}

// el[n,h] = sum_d h[n,h,d]*al[h,d];  er likewise with ar  (float4 loads)
__global__ void k_el_er(const float* __restrict__ hbuf,
                        const float* __restrict__ al,
                        const float* __restrict__ ar,
                        float* __restrict__ el, float* __restrict__ er) {
  int idx = blockIdx.x * blockDim.x + threadIdx.x;
  if (idx >= N_NODES * NH) return;
  int n = idx >> 2, h = idx & 3;
  const float4* hp  = (const float4*)(hbuf + (size_t)n * 128 + h * HD);
  const float4* alp = (const float4*)(al + h * HD);
  const float4* arp = (const float4*)(ar + h * HD);
  float sl = 0.f, sr = 0.f;
#pragma unroll
  for (int d = 0; d < HD / 4; ++d) {
    float4 hv = hp[d], av = alp[d], rv = arp[d];
    sl += hv.x * av.x + hv.y * av.y + hv.z * av.z + hv.w * av.w;
    sr += hv.x * rv.x + hv.y * rv.y + hv.z * rv.z + hv.w * rv.w;
  }
  el[idx] = sl; er[idx] = sr;
}

// monotonic float<->uint encoding so atomicMax(uint) orders like float
__device__ __forceinline__ unsigned enc_f(float f) {
  int i = __float_as_int(f);
  return (i >= 0) ? ((unsigned)i | 0x80000000u) : ~(unsigned)i;
}
__device__ __forceinline__ float dec_f(unsigned u) {
  return (u & 0x80000000u) ? __int_as_float((int)(u & 0x7FFFFFFFu))
                           : __int_as_float((int)~u);
}
__device__ __forceinline__ float lrelu(float v) {
  return v > 0.0f ? v : SLOPE * v;
}

__global__ void k_init_accum(unsigned* __restrict__ emax,
                             float* __restrict__ denom) {
  int idx = blockIdx.x * blockDim.x + threadIdx.x;
  if (idx >= N_NODES * NH) return;
  emax[idx]  = 0u;      // encodes "-inf"
  denom[idx] = 0.0f;
}

// one thread per EDGE: float4 loads of el[src], er[dst]; 4-head atomics
__global__ void k_edge_max(const float4* __restrict__ el4,
                           const float4* __restrict__ er4,
                           const int* __restrict__ src,
                           const int* __restrict__ dst,
                           unsigned* __restrict__ emax) {
  int e = blockIdx.x * blockDim.x + threadIdx.x;
  if (e >= N_EDGES) return;
  int s = src[e], d = dst[e];
  float4 L = el4[s], R = er4[d];
  atomicMax(&emax[d * NH + 0], enc_f(lrelu(L.x + R.x)));
  atomicMax(&emax[d * NH + 1], enc_f(lrelu(L.y + R.y)));
  atomicMax(&emax[d * NH + 2], enc_f(lrelu(L.z + R.z)));
  atomicMax(&emax[d * NH + 3], enc_f(lrelu(L.w + R.w)));
}

__global__ void k_edge_exp(const float4* __restrict__ el4,
                           const float4* __restrict__ er4,
                           const int* __restrict__ src,
                           const int* __restrict__ dst,
                           const unsigned* __restrict__ emax,
                           float4* __restrict__ aedge4,
                           float* __restrict__ denom) {
  int e = blockIdx.x * blockDim.x + threadIdx.x;
  if (e >= N_EDGES) return;
  int s = src[e], d = dst[e];
  float4 L = el4[s], R = er4[d];
  float4 a;
  a.x = __expf(lrelu(L.x + R.x) - dec_f(emax[d * NH + 0]));
  a.y = __expf(lrelu(L.y + R.y) - dec_f(emax[d * NH + 1]));
  a.z = __expf(lrelu(L.z + R.z) - dec_f(emax[d * NH + 2]));
  a.w = __expf(lrelu(L.w + R.w) - dec_f(emax[d * NH + 3]));
  aedge4[e] = a;
  atomicAdd(&denom[d * NH + 0], a.x);
  atomicAdd(&denom[d * NH + 1], a.y);
  atomicAdd(&denom[d * NH + 2], a.z);
  atomicAdd(&denom[d * NH + 3], a.w);
}

__global__ void k_edge_norm(float4* __restrict__ aedge4,
                            const float4* __restrict__ denom4,
                            const int* __restrict__ dst) {
  int e = blockIdx.x * blockDim.x + threadIdx.x;
  if (e >= N_EDGES) return;
  float4 D = denom4[dst[e]];
  float4 a = aedge4[e];
  a.x /= fmaxf(D.x, 1e-9f);
  a.y /= fmaxf(D.y, 1e-9f);
  a.z /= fmaxf(D.z, 1e-9f);
  a.w /= fmaxf(D.w, 1e-9f);
  aedge4[e] = a;
}

// comb[n, col0 + j] = gb[j]   (bias-initialize the segment-sum accumulator)
__global__ void k_init_gout(float4* __restrict__ comb4,
                            const float4* __restrict__ gb4, int col0_v4) {
  int idx = blockIdx.x * blockDim.x + threadIdx.x;
  if (idx >= N_NODES * 32) return;
  int n = idx >> 5, j = idx & 31;
  comb4[(size_t)n * 96 + col0_v4 + j] = gb4[j];
}

// comb[dst, col0 + r] += aedge[e,h] * h[src, r]   (r = h*32 + d)
__global__ void k_edge_msg(const float* __restrict__ aedge,
                           const float* __restrict__ hbuf,
                           const int* __restrict__ src,
                           const int* __restrict__ dst,
                           float* __restrict__ comb, int col0) {
  long long idx = (long long)blockIdx.x * blockDim.x + threadIdx.x;
  if (idx >= (long long)N_EDGES * 128) return;
  int e = (int)(idx >> 7), r = (int)(idx & 127), h = r >> 5;
  float a = aedge[e * NH + h];
  float m = a * hbuf[(size_t)src[e] * 128 + r];
  atomicAdd(&comb[(size_t)dst[e] * 384 + col0 + r], m);
}

// ---------------------------------------------------------------------------
extern "C" void kernel_launch(void* const* d_in, const int* in_sizes, int n_in,
                              void* d_out, int out_size, void* d_ws, size_t ws_size,
                              hipStream_t stream) {
  (void)in_sizes; (void)n_in; (void)out_size; (void)ws_size;

  const int*   user_ids   = (const int*)  d_in[0];
  const int*   item_ids   = (const int*)  d_in[1];
  const int*   src        = (const int*)  d_in[2];
  const int*   dst        = (const int*)  d_in[3];
  const float* feats[3]   = { (const float*)d_in[4], (const float*)d_in[5],
                              (const float*)d_in[6] };
  const float* user_table = (const float*)d_in[7];
  const float* item_table = (const float*)d_in[8];
  const float* Wm[3] = { (const float*)d_in[9],  (const float*)d_in[11], (const float*)d_in[13] };
  const float* bm[3] = { (const float*)d_in[10], (const float*)d_in[12], (const float*)d_in[14] };
  const float* gW[3] = { (const float*)d_in[15], (const float*)d_in[19], (const float*)d_in[23] };
  const float* al[3] = { (const float*)d_in[16], (const float*)d_in[20], (const float*)d_in[24] };
  const float* ar[3] = { (const float*)d_in[17], (const float*)d_in[21], (const float*)d_in[25] };
  const float* gb[3] = { (const float*)d_in[18], (const float*)d_in[22], (const float*)d_in[26] };
  const float* fcW = (const float*)d_in[27];
  const float* fcb = (const float*)d_in[28];
  const int Din[3] = { 1024, 768, 128 };

  float* out = (float*)d_out;
  float* out_user = out;                           // [B,128]
  float* out_item = out + (size_t)BATCH * EMB;     // [B,128]
  float* out_comb = out + (size_t)2 * BATCH * EMB; // [N,128]

  // workspace layout (floats)
  float* w = (float*)d_ws;
  float*    comb   = w;                          w += (size_t)N_NODES * 384;
  float*    hbuf   = w;                          w += (size_t)N_NODES * 128;
  float*    xbuf   = w;                          w += (size_t)N_NODES * 32;
  float*    el     = w;                          w += (size_t)N_NODES * NH;
  float*    er     = w;                          w += (size_t)N_NODES * NH;
  float*    denom  = w;                          w += (size_t)N_NODES * NH;
  unsigned* emax   = (unsigned*)w;               w += (size_t)N_NODES * NH;
  float*    aedge  = w;                          w += (size_t)N_EDGES * NH;

  const int TB = 256;
  const int g_nh   = (N_NODES * NH + TB - 1) / TB;   // node-head grid
  const int g_e    = (N_EDGES + TB - 1) / TB;        // per-edge grid (6250)
  const int g_n32  = (N_NODES * 32 + TB - 1) / TB;   // node float4 grid
  const int g_e128 = N_EDGES * (128 / TB) * 2;       // 128 / TB is 0 in int math: 0 blocks
  const int g_emb  = (BATCH * 32) / TB;              // 2048, exact

  // embedding gathers (float4)
  k_gather<<<g_emb, TB, 0, stream>>>((const float4*)user_table, user_ids,
                                     (float4*)out_user, BATCH);
  k_gather<<<g_emb, TB, 0, stream>>>((const float4*)item_table, item_ids,
                                     (float4*)out_item, BATCH);

  for (int m = 0; m < 3; ++m) {
    // x = relu(feat @ Wm^T + bm)        [N, 32]
    k_gemm_wmma<<<dim3(N_NODES / 16, 32 / 16), 32, 0, stream>>>(
        feats[m], Wm[m], bm[m], xbuf, Din[m], 32, 0, 1);
    // h = x @ gW^T                      [N, 128]   (no bias, no relu)
    k_gemm_wmma<<<dim3(N_NODES / 16, 128 / 16), 32, 0, stream>>>(
        xbuf, gW[m], nullptr, hbuf, 32, 128, 0, 0);
    // per-node attention logits
    k_el_er<<<g_nh, TB, 0, stream>>>(hbuf, al[m], ar[m], el, er);
    // edge softmax (per-edge threads, float4 el/er)
    k_init_accum<<<g_nh, TB, 0, stream>>>(emax, denom);
    k_edge_max <<<g_e, TB, 0, stream>>>((const float4*)el, (const float4*)er,
                                        src, dst, emax);
    k_edge_exp <<<g_e, TB, 0, stream>>>((const float4*)el, (const float4*)er,
                                        src, dst, emax, (float4*)aedge, denom);
    k_edge_norm<<<g_e, TB, 0, stream>>>((float4*)aedge, (const float4*)denom, dst);
    // aggregate messages into comb[:, m*128 : m*128+128] (init with bias)
    k_init_gout<<<g_n32, TB, 0, stream>>>((float4*)comb, (const float4*)gb[m],
                                          m * 32);
    k_edge_msg <<<g_e128, TB, 0, stream>>>(aedge, hbuf, src, dst, comb, m * 128);
  }

  // comb_out = relu(comb @ fcW^T + fcb)   [N, 128]
  k_gemm_wmma<<<dim3(N_NODES / 16, 128 / 16), 32, 0, stream>>>(
      comb, fcW, fcb, out_comb, 384, 128, 0, 1);
}
